// RetroModel_22600117911817
// MI455X (gfx1250) — matrix operand order repaired and checked by reading, not compiled, not run on hardware
//
#include <hip/hip_runtime.h>
#include <stdint.h>

#define DEV __device__ __forceinline__

constexpr int cBt = 4, cS = 2048, cD = 1024, cH = 16, cDK = 64;
constexpr int cCL = 64, cC = 32, cNN = 2, cNL = 128;
constexpr int cHD = cH * cDK;          // 1024
constexpr int cJJ = cNN * cNL;         // 256
constexpr float cSCALE = 0.125f;       // 1/sqrt(64)

typedef __attribute__((ext_vector_type(16))) __bf16 v16bf;
typedef __attribute__((ext_vector_type(8)))  float  v8f;

union AFrag { v16bf v; uint32_t u[8]; };
union BFrag { v16bf v; uint4 q[2]; };

DEV uint16_t f2bf(float f) {
  union { float f; uint32_t u; } x; x.f = f;
  uint32_t r = x.u + 0x7FFFu + ((x.u >> 16) & 1u);
  return (uint16_t)(r >> 16);
}

// CDNA5 async global->LDS copy (16B per lane), tracked by ASYNCcnt
DEV void async_ld_b128(const uint16_t* lds, const uint16_t* g) {
  asm volatile("global_load_async_to_lds_b128 %0, %1, off"
               :: "v"((uint32_t)(uintptr_t)lds), "v"(g)
               : "memory");
}
DEV void wait_async0() {
  asm volatile("s_wait_asynccnt 0x0" ::: "memory");
}

// A fragment: 16x32 bf16, lane = M%16, K interleaved per ISA table
DEV void load_a_frag(AFrag& a, const uint16_t* lds, int rowBase, int stride, int kk, int lane) {
  const uint16_t* p = lds + (rowBase + (lane & 15)) * stride + kk;
  const int kh = (lane >> 4) << 3;     // lanes>=16 hold K+8
#pragma unroll
  for (int g = 0; g < 8; ++g) {
    int kp = ((g >= 4) ? 16 : 0) + kh + ((g & 3) << 1);
    a.u[g] = *(const uint32_t*)(p + kp);
  }
}
// B fragment: 32x16 bf16 from [N][K]-major LDS; lane holds column N, K interleaved per ISA table
DEV void load_b_frag(BFrag& b, const uint16_t* ldsT, int colBase, int stride, int kk, int lane) {
  const uint16_t* p = ldsT + (colBase + (lane & 15)) * stride + kk + ((lane >> 4) << 3);
  b.q[0] = *(const uint4*)(p);
  b.q[1] = *(const uint4*)(p + 16);
}
DEV v8f wmma_bf16(const AFrag& a, const BFrag& b, v8f c) {
  return __builtin_amdgcn_wmma_f32_16x16x32_bf16(false, a.v, false, b.v, (short)0, c, false, false);
}

// ---------------- elementwise f32 -> bf16 (8 elems/thread) ----------------
__global__ __launch_bounds__(256) void cvt_bf16_kernel(const float* __restrict__ src,
                                                       uint16_t* __restrict__ dst, int n8) {
  int idx = blockIdx.x * 256 + threadIdx.x;
  if (idx >= n8) return;
  const float4* s4 = (const float4*)src;
  float4 a = s4[idx * 2], b = s4[idx * 2 + 1];
  union { uint16_t h[8]; uint4 q; } o;
  o.h[0] = f2bf(a.x); o.h[1] = f2bf(a.y); o.h[2] = f2bf(a.z); o.h[3] = f2bf(a.w);
  o.h[4] = f2bf(b.x); o.h[5] = f2bf(b.y); o.h[6] = f2bf(b.z); o.h[7] = f2bf(b.w);
  ((uint4*)dst)[idx] = o.q;
}

// -------- weight convert + transpose: W[K][N] f32 -> Wt[N][K] bf16 --------
__global__ __launch_bounds__(256) void cvt_wt_kernel(const float* __restrict__ W,
                                                     uint16_t* __restrict__ Wt,
                                                     int K, int Nn) {
  __shared__ uint16_t T[64 * 72];
  const int tid = threadIdx.x;
  const int kt = blockIdx.x * 64, nt = blockIdx.y * 64;
#pragma unroll
  for (int i = 0; i < 16; ++i) {
    int e = i * 256 + tid; int r = e >> 6, c = e & 63;          // r: k, c: n
    T[r * 72 + c] = f2bf(W[(size_t)(kt + r) * Nn + nt + c]);
  }
  __syncthreads();
#pragma unroll
  for (int i = 0; i < 2; ++i) {
    int u = i * 256 + tid;                                      // 512 uint4 outputs
    int nl = u >> 3, kg = (u & 7) * 8;
    union { uint16_t h[8]; uint4 q; } o;
#pragma unroll
    for (int t = 0; t < 8; ++t) o.h[t] = T[(kg + t) * 72 + nl];
    *(uint4*)&Wt[(size_t)(nt + nl) * K + kt + kg] = o.q;
  }
}

// ---------------- LayerNorm + shift + pad, fp32 -> bf16 ----------------
__global__ __launch_bounds__(256) void ln_kernel(const float* __restrict__ h,
                                                 const float* __restrict__ lnw,
                                                 const float* __restrict__ lnb,
                                                 uint16_t* __restrict__ hb) {
  const int row = blockIdx.x;          // 0 .. B*2048-1
  const int bb = row >> 11, s0 = row & 2047;
  const int tid = threadIdx.x;
  uint16_t* outp = hb + (size_t)row * cD;
  if (s0 > cS - cCL) {                 // padded rows (1985..2047)
    uint2 z; z.x = 0u; z.y = 0u;
    ((uint2*)outp)[tid] = z;
    return;
  }
  const float4* src4 = (const float4*)(h + ((size_t)bb * cS + s0 + cCL - 1) * cD);
  float4 x = src4[tid];
  float s = x.x + x.y + x.z + x.w;
  float sq = x.x * x.x + x.y * x.y + x.z * x.z + x.w * x.w;
#pragma unroll
  for (int m = 1; m < 32; m <<= 1) { s += __shfl_xor(s, m, 32); sq += __shfl_xor(sq, m, 32); }
  __shared__ float rs[8], rq[8];
  __shared__ float muS, rstdS;
  if ((tid & 31) == 0) { rs[tid >> 5] = s; rq[tid >> 5] = sq; }
  __syncthreads();
  if (tid == 0) {
    float S1 = 0.f, S2 = 0.f;
#pragma unroll
    for (int i = 0; i < 8; ++i) { S1 += rs[i]; S2 += rq[i]; }
    float mu = S1 * (1.f / cD);
    float var = S2 * (1.f / cD) - mu * mu;
    muS = mu; rstdS = rsqrtf(var + 1e-5f);
  }
  __syncthreads();
  float mu = muS, r = rstdS;
  float4 g4 = ((const float4*)lnw)[tid];
  float4 b4 = ((const float4*)lnb)[tid];
  union { uint16_t h[4]; uint2 q; } o;
  o.h[0] = f2bf((x.x - mu) * r * g4.x + b4.x);
  o.h[1] = f2bf((x.y - mu) * r * g4.y + b4.y);
  o.h[2] = f2bf((x.z - mu) * r * g4.z + b4.z);
  o.h[3] = f2bf((x.w - mu) * r * g4.w + b4.w);
  ((uint2*)outp)[tid] = o.q;
}

// ---------------- residual prefix rows [0, CL-1) ----------------
__global__ void prefix_kernel(const float* __restrict__ h, float* __restrict__ out) {
  int idx = blockIdx.x * 256 + threadIdx.x;           // float4 index
  if (idx >= cBt * (cCL - 1) * (cD / 4)) return;
  int d4 = idx & (cD / 4 - 1);
  int t = idx >> 8;
  int s = t % (cCL - 1), bb = t / (cCL - 1);
  size_t o = ((size_t)bb * cS + s) * (cD / 4) + d4;
  ((float4*)out)[o] = ((const float4*)h)[o];
}

// ------- 128x128x32-tile bf16 WMMA GEMM, async double-buffered LDS -------
// A[M][K] bf16, Bt[N][K] bf16 (pre-transposed weights)
// !FINAL: outBf[m,n] = bf16(A@W + bias), coalesced b128 stores via LDS bounce
//  FINAL: out[b, s0+CL-1, n] = acc + bias[n] + resid  (skip cropped rows)
template<bool FINAL>
__global__ __launch_bounds__(256) void gemm_kernel(
    const uint16_t* __restrict__ A, const uint16_t* __restrict__ Bt,
    const float* __restrict__ bias,
    uint16_t* __restrict__ outBf, float* __restrict__ outF, const float* __restrict__ resid,
    int K, int Nn) {
  constexpr int LD = 40;
  constexpr int TILE = 128 * LD;                           // elems per buffer
  __shared__ __align__(16) uint16_t smemG[4 * TILE];       // 40960 B
  const int tid = threadIdx.x, lane = tid & 31, wave = tid >> 5;
  const int wm = wave >> 1, wn = wave & 1;
  const size_t bm = (size_t)blockIdx.x * 128, bn = (size_t)blockIdx.y * 128;

  // per-thread staging coords: two b128 per matrix per tile
  const int r0 = tid >> 2, c0 = (tid & 3) * 8;         // rows 0..63
  const int r1 = 64 + r0;                              // rows 64..127
  const uint16_t* gA = A + bm * (size_t)K;
  const uint16_t* gB = Bt + bn * (size_t)K;

  v8f acc[2][4];
#pragma unroll
  for (int i = 0; i < 2; ++i)
#pragma unroll
    for (int j = 0; j < 4; ++j)
#pragma unroll
      for (int vv = 0; vv < 8; ++vv) acc[i][j][vv] = 0.f;

  // async-stage tile 0 (buffers computed by offset, no LDS pointer arrays)
  async_ld_b128(&smemG[r0 * LD + c0], &gA[(size_t)r0 * K + c0]);
  async_ld_b128(&smemG[r1 * LD + c0], &gA[(size_t)r1 * K + c0]);
  async_ld_b128(&smemG[2 * TILE + r0 * LD + c0], &gB[(size_t)r0 * K + c0]);
  async_ld_b128(&smemG[2 * TILE + r1 * LD + c0], &gB[(size_t)r1 * K + c0]);
  wait_async0();
  __syncthreads();

  int cur = 0;
  for (int k0 = 0; k0 < K; k0 += 32) {
    const bool more = (k0 + 32) < K;
    const uint16_t* sAc = smemG + cur * TILE;
    const uint16_t* sBc = smemG + 2 * TILE + cur * TILE;
    if (more) {                                        // async-prefetch next tile
      const int nxt = cur ^ 1;
      uint16_t* sAn = smemG + nxt * TILE;
      uint16_t* sBn = smemG + 2 * TILE + nxt * TILE;
      async_ld_b128(&sAn[r0 * LD + c0], &gA[(size_t)r0 * K + k0 + 32 + c0]);
      async_ld_b128(&sAn[r1 * LD + c0], &gA[(size_t)r1 * K + k0 + 32 + c0]);
      async_ld_b128(&sBn[r0 * LD + c0], &gB[(size_t)r0 * K + k0 + 32 + c0]);
      async_ld_b128(&sBn[r1 * LD + c0], &gB[(size_t)r1 * K + k0 + 32 + c0]);
    }
    AFrag af[2]; BFrag bfr[4];
#pragma unroll
    for (int im = 0; im < 2; ++im) load_a_frag(af[im], sAc, wm * 32 + im * 16, LD, 0, lane);
#pragma unroll
    for (int jn = 0; jn < 4; ++jn) load_b_frag(bfr[jn], sBc, wn * 64 + jn * 16, LD, 0, lane);
#pragma unroll
    for (int im = 0; im < 2; ++im)
#pragma unroll
      for (int jn = 0; jn < 4; ++jn)
        acc[im][jn] = wmma_bf16(af[im], bfr[jn], acc[im][jn]);
    wait_async0();
    __syncthreads();
    cur ^= 1;
  }

  const int half8 = (lane >> 4) << 3;
  if constexpr (!FINAL) {
    // bounce through LDS -> coalesced b128 stores
    uint16_t* sOut = smemG;                             // [128][136]
#pragma unroll
    for (int im = 0; im < 2; ++im)
#pragma unroll
      for (int jn = 0; jn < 4; ++jn) {
        int n_l = wn * 64 + jn * 16 + (lane & 15);
        float bv = bias[bn + n_l];
#pragma unroll
        for (int vv = 0; vv < 8; ++vv) {
          int r_l = wm * 32 + im * 16 + half8 + vv;
          sOut[r_l * 136 + n_l] = f2bf(acc[im][jn][vv] + bv);
        }
      }
    __syncthreads();
    const int orow = tid >> 1, oseg = (tid & 1) * 64;
#pragma unroll
    for (int j = 0; j < 8; ++j) {
      uint4 d = *(const uint4*)&sOut[orow * 136 + oseg + j * 8];
      *(uint4*)&outBf[(bm + orow) * (size_t)Nn + bn + oseg + j * 8] = d;
    }
  } else {
    // two 64-row halves of f32 through LDS -> float4 residual add + stores
    float* sOutF = (float*)smemG;                       // [64][132]
#pragma unroll
    for (int hf = 0; hf < 2; ++hf) {
      __syncthreads();
      if ((wm >> 1) == hf) {
        int wml = wm & 1;
#pragma unroll
        for (int im = 0; im < 2; ++im)
#pragma unroll
          for (int jn = 0; jn < 4; ++jn) {
            int n_l = wn * 64 + jn * 16 + (lane & 15);
            float bv = bias[bn + n_l];
#pragma unroll
            for (int vv = 0; vv < 8; ++vv) {
              int r_l = wml * 32 + im * 16 + half8 + vv;
              sOutF[r_l * 132 + n_l] = acc[im][jn][vv] + bv;
            }
          }
      }
      __syncthreads();
#pragma unroll
      for (int i = 0; i < 8; ++i) {
        int u = i * 256 + tid;
        int r_l = u >> 5, c4 = (u & 31) * 4;
        size_t gr = bm + hf * 64 + r_l;
        int s0 = (int)(gr & 2047);
        if (s0 <= cS - cCL) {
          size_t o = ((gr >> 11) * cS + (size_t)(s0 + cCL - 1)) * (size_t)Nn + bn + c4;
          float4 rv = *(const float4*)&resid[o];
          float4 sv = *(const float4*)&sOutF[r_l * 132 + c4];
          float4 w;
          w.x = sv.x + rv.x; w.y = sv.y + rv.y; w.z = sv.z + rv.z; w.w = sv.w + rv.w;
          *(float4*)&outF[o] = w;
        }
      }
    }
  }
}

// ---------------- fused attention per (b*c, head): 64x256 scores ----------------
__global__ __launch_bounds__(256) void attn_kernel(const uint16_t* __restrict__ qg,
                                                   const uint16_t* __restrict__ kg,
                                                   const uint16_t* __restrict__ vg,
                                                   uint16_t* __restrict__ og) {
  const int hh = blockIdx.x & (cH - 1);
  const int bc = blockIdx.x >> 4;                   // b*C + c, 0..127
  const int tid = threadIdx.x, lane = tid & 31, wave = tid >> 5;
  const int wm = wave >> 1, wn = wave & 1;

  __shared__ __align__(16) uint16_t smem[23040];    // 46080 B, phase-aliased
  uint16_t* sQ = smem;                              // phase A: [64][72]
  uint16_t* sK = smem + 64 * 72;                    // phase A: [256][72]
  uint16_t* sP = smem;                              // phase C: [64][72] per chunk
  uint16_t* sVT = smem + 64 * 72;                   // phase C: [64 d][72 j']
  __shared__ float rMax[2][64], rSum[2][64], rowMaxSh[64], rowSumSh[64];

  const uint16_t* qp = qg + (size_t)bc * cCL * cHD + hh * cDK;
  const uint16_t* kptr = kg + (size_t)bc * cJJ * cHD + hh * cDK;
#pragma unroll
  for (int i = 0; i < 2; ++i) {                     // Q: async 512 b128
    int u = i * 256 + tid; int r = u >> 3, c8 = (u & 7) * 8;
    async_ld_b128(&sQ[r * 72 + c8], &qp[(size_t)r * cHD + c8]);
  }
#pragma unroll
  for (int i = 0; i < 8; ++i) {                     // K: async 2048 b128
    int u = i * 256 + tid; int r = u >> 3, c8 = (u & 7) * 8;
    async_ld_b128(&sK[r * 72 + c8], &kptr[(size_t)r * cHD + c8]);
  }
  wait_async0();
  __syncthreads();

  // S = q @ k^T : wave covers rows wm*16..+16, cols wn*128..+128 (8 tiles)
  v8f acc[8];
#pragma unroll
  for (int j = 0; j < 8; ++j)
#pragma unroll
    for (int vv = 0; vv < 8; ++vv) acc[j][vv] = 0.f;
  for (int kk = 0; kk < cDK; kk += 32) {
    AFrag aq; load_a_frag(aq, sQ, wm * 16, 72, kk, lane);
#pragma unroll
    for (int jn = 0; jn < 8; ++jn) {
      BFrag bk_; load_b_frag(bk_, sK, wn * 128 + jn * 16, 72, kk, lane);
      acc[jn] = wmma_bf16(aq, bk_, acc[jn]);
    }
  }

  // softmax over 256 (raw max; scale folded into exp2)
  const int rb = wm * 16 + ((lane >> 4) << 3);
  float pm[8];
#pragma unroll
  for (int vv = 0; vv < 8; ++vv) {
    float m0 = acc[0][vv];
#pragma unroll
    for (int jn = 1; jn < 8; ++jn) m0 = fmaxf(m0, acc[jn][vv]);
    pm[vv] = m0;
  }
#pragma unroll
  for (int msk = 1; msk < 16; msk <<= 1)
#pragma unroll
    for (int vv = 0; vv < 8; ++vv) pm[vv] = fmaxf(pm[vv], __shfl_xor(pm[vv], msk, 32));
  if ((lane & 15) == 0)
#pragma unroll
    for (int vv = 0; vv < 8; ++vv) rMax[wn][rb + vv] = pm[vv];
  __syncthreads();
  if (tid < 64) rowMaxSh[tid] = fmaxf(rMax[0][tid], rMax[1][tid]);
  __syncthreads();
  const float c0 = cSCALE * 1.4426950408889634f;
  float ps[8];
#pragma unroll
  for (int vv = 0; vv < 8; ++vv) {
    float rm = rowMaxSh[rb + vv]; float ssum = 0.f;
#pragma unroll
    for (int jn = 0; jn < 8; ++jn) {
      float p = exp2f((acc[jn][vv] - rm) * c0);
      acc[jn][vv] = p; ssum += p;
    }
    ps[vv] = ssum;
  }
#pragma unroll
  for (int msk = 1; msk < 16; msk <<= 1)
#pragma unroll
    for (int vv = 0; vv < 8; ++vv) ps[vv] += __shfl_xor(ps[vv], msk, 32);
  if ((lane & 15) == 0)
#pragma unroll
    for (int vv = 0; vv < 8; ++vv) rSum[wn][rb + vv] = ps[vv];
  __syncthreads();
  if (tid < 64) rowSumSh[tid] = rSum[0][tid] + rSum[1][tid];

  // O = P @ V in 4 chunks of 64 keys, LDS reused for P and V^T
  v8f oacc[2];
#pragma unroll
  for (int t = 0; t < 2; ++t)
#pragma unroll
    for (int vv = 0; vv < 8; ++vv) oacc[t][vv] = 0.f;
  const int on0 = (wave & 1) * 2;
  const uint16_t* vptr = vg + (size_t)bc * cJJ * cHD + hh * cDK;

  for (int g = 0; g < 4; ++g) {
    __syncthreads();
    if (wn == (g >> 1)) {                            // owning waves write exp'd P (bf16)
#pragma unroll
      for (int t = 0; t < 4; ++t) {
        int jn = (g & 1) * 4 + t;
        int cl_ = t * 16 + (lane & 15);
#pragma unroll
        for (int vv = 0; vv < 8; ++vv)
          sP[(rb + vv) * 72 + cl_] = f2bf(acc[jn][vv]);
      }
    }
#pragma unroll
    for (int i = 0; i < 2; ++i) {                    // stage V chunk transposed [d][j']
      int u = i * 256 + tid; int r = u >> 3, c8 = (u & 7) * 8;
      union { uint4 q; uint16_t h[8]; } vld;
      vld.q = *(const uint4*)&vptr[(size_t)(g * 64 + r) * cHD + c8];
#pragma unroll
      for (int t = 0; t < 8; ++t) sVT[(c8 + t) * 72 + r] = vld.h[t];
    }
    __syncthreads();
    for (int kk = 0; kk < 64; kk += 32) {
      AFrag ap; load_a_frag(ap, sP, wm * 16, 72, kk, lane);
#pragma unroll
      for (int t = 0; t < 2; ++t) {
        BFrag bv_; load_b_frag(bv_, sVT, (on0 + t) * 16, 72, kk, lane);
        oacc[t] = wmma_bf16(ap, bv_, oacc[t]);
      }
    }
  }

  // normalize and store bf16 o [B*C*CL][H*DK]
  uint16_t* op = og + (size_t)bc * cCL * cHD + hh * cDK;
#pragma unroll
  for (int t = 0; t < 2; ++t) {
    int d = (on0 + t) * 16 + (lane & 15);
#pragma unroll
    for (int vv = 0; vv < 8; ++vv) {
      int row = rb + vv;
      op[(size_t)row * cHD + d] = f2bf(oacc[t][vv] / rowSumSh[row]);
    }
  }
}

extern "C" void kernel_launch(void* const* d_in, const int* in_sizes, int n_in,
                              void* d_out, int out_size, void* d_ws, size_t ws_size,
                              hipStream_t stream) {
  (void)in_sizes; (void)n_in; (void)out_size; (void)ws_size;
  const float* h   = (const float*)d_in[0];
  const float* e   = (const float*)d_in[1];
  const float* lng = (const float*)d_in[2];
  const float* lnb = (const float*)d_in[3];
  const float* Wq  = (const float*)d_in[4];
  const float* bq  = (const float*)d_in[5];
  const float* Wk  = (const float*)d_in[6];
  const float* bk  = (const float*)d_in[7];
  const float* Wv  = (const float*)d_in[8];
  const float* bv  = (const float*)d_in[9];
  const float* Wo  = (const float*)d_in[10];
  const float* bo  = (const float*)d_in[11];
  float* out = (float*)d_out;

  const size_t MQ = (size_t)cBt * cC * cCL;          // 8192
  const size_t MK = (size_t)cBt * cC * cJJ;          // 32768
  uint16_t* ws = (uint16_t*)d_ws;
  uint16_t* hbuf = ws;                               // 8192x1024
  uint16_t* qbuf = hbuf + MQ * cHD;
  uint16_t* kbuf = qbuf + MQ * cHD;                  // 32768x1024
  uint16_t* vbuf = kbuf + MK * cHD;
  uint16_t* obuf = vbuf + MK * cHD;                  // 8192x1024
  uint16_t* ebuf = obuf + MQ * cHD;                  // 32768x1024 bf16 e
  uint16_t* wqt  = ebuf + MK * cHD;                  // 1024x1024 each, transposed
  uint16_t* wkt  = wqt + (size_t)cD * cHD;
  uint16_t* wvt  = wkt + (size_t)cD * cHD;
  uint16_t* wot  = wvt + (size_t)cD * cHD;

  // one-time conversions (bf16, weights transposed to [N][K])
  cvt_bf16_kernel<<<(int)(MK * cD / 8 / 256), 256, 0, stream>>>(e, ebuf, (int)(MK * cD / 8));
  cvt_wt_kernel<<<dim3(16, 16), 256, 0, stream>>>(Wq, wqt, cD, cHD);
  cvt_wt_kernel<<<dim3(16, 16), 256, 0, stream>>>(Wk, wkt, cD, cHD);
  cvt_wt_kernel<<<dim3(16, 16), 256, 0, stream>>>(Wv, wvt, cD, cHD);
  cvt_wt_kernel<<<dim3(16, 16), 256, 0, stream>>>(Wo, wot, cHD, cD);

  ln_kernel<<<cBt * cS, 256, 0, stream>>>(h, lng, lnb, hbuf);
  prefix_kernel<<<(cBt * (cCL - 1) * (cD / 4) + 255) / 256, 256, 0, stream>>>(h, out);

  gemm_kernel<false><<<dim3(MQ / 128, cHD / 128), 256, 0, stream>>>(
      hbuf, wqt, bq, qbuf, nullptr, nullptr, cD, cHD);
  gemm_kernel<false><<<dim3(MK / 128, cHD / 128), 256, 0, stream>>>(
      ebuf, wkt, bk, kbuf, nullptr, nullptr, cD, cHD);
  gemm_kernel<false><<<dim3(MK / 128, cHD / 128), 256, 0, stream>>>(
      ebuf, wvt, bv, vbuf, nullptr, nullptr, cD, cHD);

  attn_kernel<<<cBt * cC * cH, 256, 0, stream>>>(qbuf, kbuf, vbuf, obuf);

  gemm_kernel<true><<<dim3(MQ / 128, cD / 128), 256, 0, stream>>>(
      obuf, wot, bo, nullptr, out, h, cHD, cD);
}
